// BackbonePointNet_16561393894097
// MI455X (gfx1250) — matrix operand order repaired and checked by reading, not compiled, not run on hardware
//
#include <hip/hip_runtime.h>
#include <hip/hip_bf16.h>
#include <stdint.h>

// ---------------------------------------------------------------------------
// CDNA5 (gfx1250) PointNet GNN: wave32 + V_WMMA_F32_16X16X32_BF16
// One node (16 incoming edges, contiguous by construction) = one 16-row tile.
// Bias is injected via the WMMA C operand (same column per lane across regs).
// ---------------------------------------------------------------------------

typedef __bf16 v16bf __attribute__((ext_vector_type(16)));
typedef float  v8f   __attribute__((ext_vector_type(8)));

union Frag16 { v16bf v; uint32_t w[8]; uint4 q[2]; };
union Bf2    { __bf16 h[2]; uint32_t w; };

// Pack an f32 [Kreal][Creal] weight matrix into bf16 WMMA B-fragments.
// Fragment (kc,nt): 32 lanes x 8 dwords; lane L holds column n=(L&15)+16*nt,
// k-halves split at lane 16; dword w holds (k, k+1) with k = kc*32+16*(L>>4)+2w.
__global__ void pack_weight_frags(const float* __restrict__ w, uint32_t* __restrict__ out,
                                  int Kreal, int Creal, int KC, int NT) {
  int total = KC * NT * 256;
  for (int idx = blockIdx.x * blockDim.x + threadIdx.x; idx < total;
       idx += gridDim.x * blockDim.x) {
    int word = idx & 7;
    int lane = (idx >> 3) & 31;
    int frag = idx >> 8;
    int nt = frag % NT;
    int kc = frag / NT;
    int n = (lane & 15) + nt * 16;
    int k = kc * 32 + ((lane >> 4) << 4) + 2 * word;
    Bf2 p;
    p.h[0] = (__bf16)((k     < Kreal && n < Creal) ? w[(size_t)k       * Creal + n] : 0.0f);
    p.h[1] = (__bf16)((k + 1 < Kreal && n < Creal) ? w[(size_t)(k + 1) * Creal + n] : 0.0f);
    out[idx] = p.w;
  }
}

__global__ void zero_f32(float* __restrict__ p, int n) {
  int i = blockIdx.x * blockDim.x + threadIdx.x;
  if (i < n) p[i] = 0.0f;
}

// ---------------------------------------------------------------------------
// One PointNet message-passing layer.
//  KPA   : padded K of GEMM1 (32 or 96)
//  CMID  : hidden width (64/128), also K of GEMM2
//  COUT  : output width (64/128)
//  SCOLS : LDS staging row width in bf16 elements = max(KPA, CMID)
// ---------------------------------------------------------------------------
template<int KPA, int CMID, int COUT, int SCOLS, bool FIRST, bool FINAL>
__global__ void pointnet_layer(
    const float* __restrict__ pos,
    const int* __restrict__ src,          // edge sources; edges of node i at [16i,16i+16)
    const __bf16* __restrict__ h_in,      // bf16 [N][64] (unused if FIRST)
    const uint32_t* __restrict__ waFrag,
    const uint32_t* __restrict__ wbFrag,
    const float* __restrict__ ba,
    const float* __restrict__ bb,
    __bf16* __restrict__ h_out,           // bf16 [N][COUT] (if !FINAL)
    float* __restrict__ pooled,           // [B][COUT]     (if FINAL)
    float* __restrict__ counts,           // [B]           (if FINAL)
    const int* __restrict__ batch,
    int nNodes)
{
  constexpr int KCA = KPA / 32;
  constexpr int NTM = CMID / 16;
  constexpr int KCB = CMID / 32;
  constexpr int NTO = COUT / 16;
  constexpr int WA_WORDS = KCA * NTM * 256;
  constexpr int WB_WORDS = KCB * NTO * 256;

  extern __shared__ char smem[];
  uint32_t* lwa = (uint32_t*)smem;
  uint32_t* lwb = lwa + WA_WORDS;
  float*    lba = (float*)(lwb + WB_WORDS);
  float*    lbb = lba + CMID;
  __bf16*   stageAll = (__bf16*)(lbb + COUT);

  const int tid = threadIdx.x;
  for (int i = tid; i < WA_WORDS; i += blockDim.x) lwa[i] = waFrag[i];
  for (int i = tid; i < WB_WORDS; i += blockDim.x) lwb[i] = wbFrag[i];
  for (int i = tid; i < CMID;     i += blockDim.x) lba[i] = ba[i];
  for (int i = tid; i < COUT;     i += blockDim.x) lbb[i] = bb[i];
  __syncthreads();

  const int lane = tid & 31;
  const int wave = tid >> 5;
  const int wavesPerBlk = blockDim.x >> 5;
  __bf16* stage = stageAll + wave * 16 * SCOLS;
  const int row  = lane & 15;
  const int half = lane >> 4;
  const int ncol = lane & 15;            // D/C-tile column owned by this lane

  for (int node = blockIdx.x * wavesPerBlk + wave; node < nNodes;
       node += gridDim.x * wavesPerBlk) {
    const int ebase = node << 4;

    // ---- stage edge-feature matrix A (16 x KPA, bf16) into LDS ----
    if constexpr (FIRST) {
      if (lane < 16) {
        int s = src[ebase + lane];
        float qx = pos[3 * node], qy = pos[3 * node + 1], qz = pos[3 * node + 2];
        float px = pos[3 * s],    py = pos[3 * s + 1],    pz = pos[3 * s + 2];
        __bf16 tmp[32];
        #pragma unroll
        for (int i = 0; i < 32; i++) tmp[i] = (__bf16)0.0f;
        tmp[0] = (__bf16)px;        tmp[1] = (__bf16)py;        tmp[2] = (__bf16)pz;
        tmp[3] = (__bf16)(px - qx); tmp[4] = (__bf16)(py - qy); tmp[5] = (__bf16)(pz - qz);
        uint4* drow = (uint4*)(stage + lane * SCOLS);
        const uint4* srow = (const uint4*)tmp;
        #pragma unroll
        for (int i = 0; i < 4; i++) drow[i] = srow[i];
      }
    } else {
      {
        // gather h[src]: lane pair per row, 64 bf16 = 8 uint4 per row
        int r = lane >> 1;
        int s = src[ebase + r];
        const uint4* hrow = (const uint4*)(h_in + (size_t)s * 64);
        uint4* drow = (uint4*)(stage + r * SCOLS) + (lane & 1) * 4;
        #pragma unroll
        for (int i = 0; i < 4; i++) drow[i] = hrow[(lane & 1) * 4 + i];
      }
      if (lane < 16) {
        int s = src[ebase + lane];
        float qx = pos[3 * node], qy = pos[3 * node + 1], qz = pos[3 * node + 2];
        float px = pos[3 * s],    py = pos[3 * s + 1],    pz = pos[3 * s + 2];
        __bf16 tmp[32];
        #pragma unroll
        for (int i = 0; i < 32; i++) tmp[i] = (__bf16)0.0f;
        tmp[0] = (__bf16)(px - qx); tmp[1] = (__bf16)(py - qy); tmp[2] = (__bf16)(pz - qz);
        uint4* drow = (uint4*)(stage + lane * SCOLS + 64);  // cols 64..95
        const uint4* srow = (const uint4*)tmp;
        #pragma unroll
        for (int i = 0; i < 4; i++) drow[i] = srow[i];
      }
    }
    __asm__ volatile("s_wait_dscnt 0" ::: "memory");

    // ---- load A fragments for GEMM1 (16-bit A 16x32 layout) ----
    Frag16 afr[KCA];
    #pragma unroll
    for (int kc = 0; kc < KCA; kc++) {
      #pragma unroll
      for (int p = 0; p < 8; p++) {
        int k = kc * 32 + 16 * half + 2 * p;
        afr[kc].w[p] = *(const uint32_t*)(stage + row * SCOLS + k);
      }
    }

    // ---- GEMM1: (16 x KPA) x (KPA x CMID); bias pre-loaded into C ----
    v8f acc[NTM];
    #pragma unroll
    for (int nt = 0; nt < NTM; nt++) {
      float b = lba[ncol + nt * 16];
      acc[nt] = {b, b, b, b, b, b, b, b};
      #pragma unroll
      for (int kc = 0; kc < KCA; kc++) {
        Frag16 bfr;
        const uint4* bp = (const uint4*)(lwa + ((kc * NTM + nt) * 32 + lane) * 8);
        bfr.q[0] = bp[0];
        bfr.q[1] = bp[1];
        acc[nt] = __builtin_amdgcn_wmma_f32_16x16x32_bf16(
            false, afr[kc].v, false, bfr.v, (short)0, acc[nt], false, false);
      }
    }

    // ---- ReLU, restage as bf16 A (16 x CMID) ----
    #pragma unroll
    for (int nt = 0; nt < NTM; nt++) {
      int n = ncol + nt * 16;
      #pragma unroll
      for (int r = 0; r < 8; r++) {
        int m = r + 8 * half;
        float v = acc[nt][r];
        v = v > 0.0f ? v : 0.0f;
        stage[m * SCOLS + n] = (__bf16)v;
      }
    }
    __asm__ volatile("s_wait_dscnt 0" ::: "memory");

    // ---- A fragments for GEMM2 ----
    Frag16 afr2[KCB];
    #pragma unroll
    for (int kc = 0; kc < KCB; kc++) {
      #pragma unroll
      for (int p = 0; p < 8; p++) {
        int k = kc * 32 + 16 * half + 2 * p;
        afr2[kc].w[p] = *(const uint32_t*)(stage + row * SCOLS + k);
      }
    }

    // ---- GEMM2: (16 x CMID) x (CMID x COUT); bias pre-loaded into C ----
    v8f acc2[NTO];
    #pragma unroll
    for (int nt = 0; nt < NTO; nt++) {
      float b = lbb[ncol + nt * 16];
      acc2[nt] = {b, b, b, b, b, b, b, b};
      #pragma unroll
      for (int kc = 0; kc < KCB; kc++) {
        Frag16 bfr;
        const uint4* bp = (const uint4*)(lwb + ((kc * NTO + nt) * 32 + lane) * 8);
        bfr.q[0] = bp[0];
        bfr.q[1] = bp[1];
        acc2[nt] = __builtin_amdgcn_wmma_f32_16x16x32_bf16(
            false, afr2[kc].v, false, bfr.v, (short)0, acc2[nt], false, false);
      }
    }

    // ---- segment-max over the 16 edge rows + outer ReLU ----
    #pragma unroll
    for (int nt = 0; nt < NTO; nt++) {
      float m = -3.0e38f;
      #pragma unroll
      for (int r = 0; r < 8; r++) {
        float v = acc2[nt][r];          // lane holds rows m = r + 8*half
        m = v > m ? v : m;
      }
      float o = __shfl_xor(m, 16, 32);  // combine the two row halves
      m = o > m ? o : m;
      m = m > 0.0f ? m : 0.0f;          // relu(agg)
      if (lane < 16) {
        int n = ncol + nt * 16;
        if constexpr (FINAL) {
          atomicAdd(&pooled[(size_t)batch[node] * COUT + n], m);
        } else {
          h_out[(size_t)node * COUT + n] = (__bf16)m;
        }
      }
    }
    if constexpr (FINAL) {
      if (lane == 0) atomicAdd(&counts[batch[node]], 1.0f);
    }
  }
}

// ---- mean pool finish + 128->64->1 regressor + sigmoid (B=64, trivial) ----
__global__ void head_kernel(const float* __restrict__ pooled,
                            const float* __restrict__ counts,
                            const float* __restrict__ wr1, const float* __restrict__ br1,
                            const float* __restrict__ wr2, const float* __restrict__ br2,
                            float* __restrict__ out, int Bn) {
  int b = blockIdx.x * blockDim.x + threadIdx.x;
  if (b >= Bn) return;
  float c = counts[b];
  c = c > 1.0f ? c : 1.0f;
  float mean[128];
  #pragma unroll
  for (int k = 0; k < 128; k++) mean[k] = pooled[b * 128 + k] / c;
  float o = br2[0];
  for (int j = 0; j < 64; j++) {
    float s = br1[j];
    #pragma unroll 8
    for (int k = 0; k < 128; k++) s += mean[k] * wr1[k * 64 + j];
    o += s * wr2[j];
  }
  out[b] = 1.0f / (1.0f + __expf(-o));
}

extern "C" void kernel_launch(void* const* d_in, const int* in_sizes, int n_in,
                              void* d_out, int out_size, void* d_ws, size_t ws_size,
                              hipStream_t stream) {
  const float* pos        = (const float*)d_in[0];
  const int*   edge_index = (const int*)d_in[1];
  const int*   batch      = (const int*)d_in[2];
  // d_in[3] = timestep: unused by the reference
  const float* w1a = (const float*)d_in[4];  const float* b1a = (const float*)d_in[5];
  const float* w1b = (const float*)d_in[6];  const float* b1b = (const float*)d_in[7];
  const float* w2a = (const float*)d_in[8];  const float* b2a = (const float*)d_in[9];
  const float* w2b = (const float*)d_in[10]; const float* b2b = (const float*)d_in[11];
  const float* w3a = (const float*)d_in[12]; const float* b3a = (const float*)d_in[13];
  const float* w3b = (const float*)d_in[14]; const float* b3b = (const float*)d_in[15];
  const float* wr1 = (const float*)d_in[16]; const float* br1 = (const float*)d_in[17];
  const float* wr2 = (const float*)d_in[18]; const float* br2 = (const float*)d_in[19];

  const int N  = in_sizes[0] / 3;
  const int Bn = out_size;               // 64
  const int* src = edge_index;           // row 0 of [2,E]; dst[e] == e/16 by construction

  // ---- workspace carve ----
  uint8_t* ws = (uint8_t*)d_ws;
  size_t o = 0;
  auto carve = [&](size_t bytes) -> uint8_t* {
    uint8_t* p = ws + o;
    o = (o + bytes + 255) & ~(size_t)255;
    return p;
  };
  __bf16*   h1  = (__bf16*)carve((size_t)N * 64 * 2);
  __bf16*   h2  = (__bf16*)carve((size_t)N * 64 * 2);
  uint32_t* fa1 = (uint32_t*)carve(1 * 4 * 256 * 4);
  uint32_t* fb1 = (uint32_t*)carve(2 * 4 * 256 * 4);
  uint32_t* fa2 = (uint32_t*)carve(3 * 4 * 256 * 4);
  uint32_t* fb2 = (uint32_t*)carve(2 * 4 * 256 * 4);
  uint32_t* fa3 = (uint32_t*)carve(3 * 8 * 256 * 4);
  uint32_t* fb3 = (uint32_t*)carve(4 * 8 * 256 * 4);
  float* pooled = (float*)carve((size_t)Bn * 128 * 4);
  float* counts = (float*)carve((size_t)Bn * 4);

  // ---- weight packing (bf16 WMMA fragments) + accumulator zeroing ----
  pack_weight_frags<<<32, 256, 0, stream>>>(w1a, fa1,   6,  64, 1, 4);
  pack_weight_frags<<<32, 256, 0, stream>>>(w1b, fb1,  64,  64, 2, 4);
  pack_weight_frags<<<32, 256, 0, stream>>>(w2a, fa2,  67,  64, 3, 4);
  pack_weight_frags<<<32, 256, 0, stream>>>(w2b, fb2,  64,  64, 2, 4);
  pack_weight_frags<<<32, 256, 0, stream>>>(w3a, fa3,  67, 128, 3, 8);
  pack_weight_frags<<<32, 256, 0, stream>>>(w3b, fb3, 128, 128, 4, 8);
  zero_f32<<<(Bn * 128 + 255) / 256, 256, 0, stream>>>(pooled, Bn * 128);
  zero_f32<<<1, 256, 0, stream>>>(counts, Bn);

  // ---- three message-passing layers; one node per wave ----
  // Layers 1-2: 8 waves/block (small LDS); layer 3: 4 waves/block (~73KB LDS).
  const size_t lds1 = (size_t)(1 * 4 * 256 + 2 * 4 * 256) * 4 + (64 + 64) * 4   + 8 * 16 * 64  * 2;
  const size_t lds2 = (size_t)(3 * 4 * 256 + 2 * 4 * 256) * 4 + (64 + 64) * 4   + 8 * 16 * 96  * 2;
  const size_t lds3 = (size_t)(3 * 8 * 256 + 4 * 8 * 256) * 4 + (128 + 128) * 4 + 4 * 16 * 128 * 2;

  pointnet_layer<32, 64, 64, 64, true, false><<<1024, 256, lds1, stream>>>(
      pos, src, nullptr, fa1, fb1, b1a, b1b, h1, nullptr, nullptr, batch, N);
  pointnet_layer<96, 64, 64, 96, false, false><<<1024, 256, lds2, stream>>>(
      pos, src, h1, fa2, fb2, b2a, b2b, h2, nullptr, nullptr, batch, N);
  pointnet_layer<96, 128, 128, 128, false, true><<<2048, 128, lds3, stream>>>(
      pos, src, h2, fa3, fb3, b3a, b3b, nullptr, pooled, counts, batch, N);

  head_kernel<<<1, 64, 0, stream>>>(pooled, counts, wr1, br1, wr2, br2,
                                    (float*)d_out, Bn);
}
